// _ClapAudioSelfAttention_3427383903070
// MI455X (gfx1250) — matrix operand order repaired and checked by reading, not compiled, not run on hardware
//
#include <hip/hip_runtime.h>

typedef __attribute__((ext_vector_type(16))) _Float16 v16h;
typedef __attribute__((ext_vector_type(8)))  _Float16 v8h;
typedef __attribute__((ext_vector_type(8)))  float    v8f;

#define DIM    512
#define NH     16
#define N_TOK  64
#define HD     32
#define NWIN   64
#define SCALE  0.17677669529663687f   // 32^-0.5

// ---- LDS layout (units: _Float16) --------------------------------------
#define QK_STRIDE 520                       // 512 + 8 pad -> conflict-free frag loads
#define VT_STRIDE 72                        // 64 + 8 pad
#define P_STRIDE  72
#define Q_OFF   0
#define K_OFF   (64 * QK_STRIDE)            // 33280
#define VT_OFF  (2 * 64 * QK_STRIDE)        // 66560
#define XP_OFF  (2 * 64 * QK_STRIDE + 512 * VT_STRIDE)   // 103424
#define LDS_HALFS (XP_OFF + 8 * 64 * P_STRIDE)           // 140288 -> 280576 B < 320 KB

static __device__ __forceinline__ v16h frag16(v8h lo, v8h hi) {
  return __builtin_shufflevector(lo, hi, 0,1,2,3,4,5,6,7,8,9,10,11,12,13,14,15);
}
// A operand (16x32 f16): lane m=l%16, K-half lh=l/16; elems 0..7 = K lh*8..+7,
// elems 8..15 = K 16+lh*8..+7  -> two contiguous 16B chunks per lane.
static __device__ __forceinline__ v16h load_fragA(const _Float16* p) {
  v8h lo = *(const v8h*)(p);
  v8h hi = *(const v8h*)(p + 16);
  return frag16(lo, hi);
}
// Contiguous-k loader (this code's order, A and B): row/col=l%16, K = lh*16 + e -> one contiguous 32B run.
static __device__ __forceinline__ v16h load_fragB(const _Float16* p) {
  v8h lo = *(const v8h*)(p);
  v8h hi = *(const v8h*)(p + 8);
  return frag16(lo, hi);
}
static __device__ __forceinline__ v8f wmma_f16(v16h a, v16h b, v8f c) {
  return __builtin_amdgcn_wmma_f32_16x16x32_f16(false, a, false, b, (short)0, c,
                                                false, false);
}

// ---- prep 1: pack [wq|wk|wv] (f32 row-major [out][in]) into B-fragment order
__global__ void pack_w_kernel(const float* __restrict__ wq,
                              const float* __restrict__ wk,
                              const float* __restrict__ wv,
                              _Float16* __restrict__ wpack) {
  int t = blockIdx.x * blockDim.x + threadIdx.x;   // 96*16 frags * 32 lanes
  int lane = t & 31;
  int frag = t >> 5;
  if (frag >= 96 * 16) return;
  int kt = frag & 15;
  int nt = frag >> 4;                               // 0..95
  int n  = nt * 16 + (lane & 15);                   // output column 0..1535
  int lh = lane >> 4;
  const float* w; int row;
  if (n < 512)       { w = wq; row = n; }
  else if (n < 1024) { w = wk; row = n - 512; }
  else               { w = wv; row = n - 1024; }
  const float* src = w + row * DIM + kt * 32 + lh * 16;
  _Float16* dst = wpack + (size_t)frag * 512 + lane * 16;
#pragma unroll
  for (int e = 0; e < 16; ++e) dst[e] = (_Float16)src[e];
}

// ---- prep 2: biasx[h][m][n] = bias_table[rel_idx[m,n]][h]
__global__ void expand_bias_kernel(const float* __restrict__ bias_table,
                                   const int* __restrict__ rel_idx,
                                   float* __restrict__ biasx) {
  int t = blockIdx.x * blockDim.x + threadIdx.x;   // 16*64*64
  int n = t & 63, m = (t >> 6) & 63, h = t >> 12;
  if (h < NH) biasx[t] = bias_table[rel_idx[m * 64 + n] * NH + h];
}

// ---- fused per-window kernel -------------------------------------------
__launch_bounds__(256, 1)
__global__ void fused_window_attn(const float* __restrict__ x,
                                  const float* __restrict__ bq,
                                  const float* __restrict__ bk,
                                  const float* __restrict__ bv,
                                  const float* __restrict__ attn_mask,
                                  const _Float16* __restrict__ wpack,
                                  const float* __restrict__ biasx,
                                  float* __restrict__ out) {
  extern __shared__ _Float16 lds[];
  _Float16* Qs = lds + Q_OFF;     // [64][520] f16
  _Float16* Ks = lds + K_OFF;     // [64][520] f16
  _Float16* Vt = lds + VT_OFF;    // [512 feat][72 tok] f16 (transposed V)
  _Float16* XP = lds + XP_OFF;    // x staging, later per-wave P scratch

  const int b    = blockIdx.x;
  const int tid  = threadIdx.x;
  const int lane = tid & 31;
  const int wid  = tid >> 5;      // 8 waves
  const int ln   = lane & 15;
  const int lh   = lane >> 4;

  // Phase 1: stage x[b] (64x512 f32) -> f16 LDS
  const float* xg = x + (size_t)b * (N_TOK * DIM);
#pragma unroll 4
  for (int i = tid; i < N_TOK * DIM; i += 256) {
    int r = i >> 9, c = i & (DIM - 1);
    XP[r * QK_STRIDE + c] = (_Float16)xg[i];
  }
  __syncthreads();

  // Phase 2: [Q|K|V] = X @ Wcat^T + bias, 96 N-tiles, wave owns 12
  for (int j = 0; j < 12; ++j) {
    const int nt   = wid * 12 + j;
    const int ccol = nt * 16 + ln;            // 0..1535
    float bias;
    if (ccol < 512)       bias = bq[ccol];
    else if (ccol < 1024) bias = bk[ccol - 512];
    else                  bias = bv[ccol - 1024];

    v8f acc[4] = {};
    const _Float16* wp = wpack + (size_t)(nt * 16) * 512 + lane * 16;
    for (int kt = 0; kt < 16; ++kt) {
      v16h bf = *(const v16h*)(wp + kt * 512);          // prepacked B frag
      const _Float16* ab = XP + kt * 32 + lh * 16;
#pragma unroll
      for (int mt = 0; mt < 4; ++mt) {
        v16h af = load_fragB(ab + (mt * 16 + ln) * QK_STRIDE);
        acc[mt] = wmma_f16(af, bf, acc[mt]);
      }
    }
#pragma unroll
    for (int mt = 0; mt < 4; ++mt)
#pragma unroll
      for (int r = 0; r < 8; ++r) {
        int row = mt * 16 + r + lh * 8;                 // C layout: M=r(+8)
        float v = acc[mt][r] + bias;
        if (ccol < 512)       Qs[row * QK_STRIDE + ccol] = (_Float16)v;
        else if (ccol < 1024) Ks[row * QK_STRIDE + (ccol - 512)] = (_Float16)v;
        else                  Vt[(ccol - 1024) * VT_STRIDE + row] = (_Float16)v;
      }
  }
  __syncthreads();

  // Phase 3: attention; wave handles heads 2*wid, 2*wid+1
  const int wb = b & (NWIN - 1);
  const float* maskw = attn_mask + (size_t)wb * N_TOK * N_TOK;
  _Float16* Pw = XP + wid * (N_TOK * P_STRIDE);

  for (int hh = 0; hh < 2; ++hh) {
    const int h  = wid * 2 + hh;
    const int hc = h * HD;

    // S = Q_h @ K_h^T  (K dim = 32 -> one WMMA per 16x16 tile)
    v8f  s[4][4];
    v16h qf[4];
#pragma unroll
    for (int mt = 0; mt < 4; ++mt)
      qf[mt] = load_fragB(Qs + (mt * 16 + ln) * QK_STRIDE + hc + lh * 16);
#pragma unroll
    for (int nt = 0; nt < 4; ++nt) {
      v16h kf = load_fragB(Ks + (nt * 16 + ln) * QK_STRIDE + hc + lh * 16);
      v8f z = {};
#pragma unroll
      for (int mt = 0; mt < 4; ++mt) s[mt][nt] = wmma_f16(qf[mt], kf, z);
    }

    // scale + rel-pos bias + mask, row softmax (rows live in 16-lane halves)
    const float* bh = biasx + h * (N_TOK * N_TOK);
#pragma unroll
    for (int mt = 0; mt < 4; ++mt) {
#pragma unroll
      for (int r = 0; r < 8; ++r) {
        const int m = mt * 16 + r + lh * 8;
        const int mrow = m * N_TOK;
        float v0 = s[mt][0][r] * SCALE + bh[mrow +  0 + ln] + maskw[mrow +  0 + ln];
        float v1 = s[mt][1][r] * SCALE + bh[mrow + 16 + ln] + maskw[mrow + 16 + ln];
        float v2 = s[mt][2][r] * SCALE + bh[mrow + 32 + ln] + maskw[mrow + 32 + ln];
        float v3 = s[mt][3][r] * SCALE + bh[mrow + 48 + ln] + maskw[mrow + 48 + ln];
        float mx = fmaxf(fmaxf(v0, v1), fmaxf(v2, v3));
#pragma unroll
        for (int d = 1; d < 16; d <<= 1) mx = fmaxf(mx, __shfl_xor(mx, d, 32));
        v0 = __expf(v0 - mx); v1 = __expf(v1 - mx);
        v2 = __expf(v2 - mx); v3 = __expf(v3 - mx);
        float sm = (v0 + v1) + (v2 + v3);
#pragma unroll
        for (int d = 1; d < 16; d <<= 1) sm += __shfl_xor(sm, d, 32);
        const float inv = __builtin_amdgcn_rcpf(sm);
        Pw[m * P_STRIDE +  0 + ln] = (_Float16)(v0 * inv);
        Pw[m * P_STRIDE + 16 + ln] = (_Float16)(v1 * inv);
        Pw[m * P_STRIDE + 32 + ln] = (_Float16)(v2 * inv);
        Pw[m * P_STRIDE + 48 + ln] = (_Float16)(v3 * inv);
      }
    }
    asm volatile("s_wait_dscnt 0" ::: "memory");   // P store -> A-frag load

    // O = P @ V_h (K dim 64 -> 2 WMMA steps; Vt rows are features -> B frags)
#pragma unroll
    for (int mt = 0; mt < 4; ++mt) {
      v8f o0 = {}, o1 = {};
#pragma unroll
      for (int k2 = 0; k2 < 2; ++k2) {
        v16h pf  = load_fragB(Pw + (mt * 16 + ln) * P_STRIDE + k2 * 32 + lh * 16);
        v16h vf0 = load_fragB(Vt + (hc +      ln) * VT_STRIDE + k2 * 32 + lh * 16);
        v16h vf1 = load_fragB(Vt + (hc + 16 + ln) * VT_STRIDE + k2 * 32 + lh * 16);
        o0 = wmma_f16(pf, vf0, o0);
        o1 = wmma_f16(pf, vf1, o1);
      }
#pragma unroll
      for (int r = 0; r < 8; ++r) {
        const int m = mt * 16 + r + lh * 8;
        float* og = out + ((size_t)b * N_TOK + m) * DIM + hc;
        og[ln]      = o0[r];
        og[16 + ln] = o1[r];
      }
    }
  }
}

extern "C" void kernel_launch(void* const* d_in, const int* in_sizes, int n_in,
                              void* d_out, int out_size, void* d_ws, size_t ws_size,
                              hipStream_t stream) {
  const float* x          = (const float*)d_in[0];
  const float* wq         = (const float*)d_in[1];
  const float* bq         = (const float*)d_in[2];
  const float* wk         = (const float*)d_in[3];
  const float* bk         = (const float*)d_in[4];
  const float* wv         = (const float*)d_in[5];
  const float* bv         = (const float*)d_in[6];
  const float* bias_table = (const float*)d_in[7];
  const float* attn_mask  = (const float*)d_in[8];
  const int*   rel_idx    = (const int*)d_in[9];
  float* out = (float*)d_out;

  _Float16* wpack = (_Float16*)d_ws;                       // 96*16*512 f16 = 1.5 MB
  float* biasx = (float*)((char*)d_ws + (size_t)96 * 16 * 512 * sizeof(_Float16));

  pack_w_kernel<<<192, 256, 0, stream>>>(wq, wk, wv, wpack);
  expand_bias_kernel<<<256, 256, 0, stream>>>(bias_table, rel_idx, biasx);

  const size_t shmem = (size_t)LDS_HALFS * sizeof(_Float16);   // 280,576 B
  fused_window_attn<<<2048, 256, shmem, stream>>>(x, bq, bk, bv, attn_mask,
                                                  wpack, biasx, out);
}
